// GraphEncoder_20418274525979
// MI455X (gfx1250) — matrix operand order repaired and checked by reading, not compiled, not run on hardware
//
#include <hip/hip_runtime.h>
#include <hip/hip_bf16.h>

#define HIDDEN   512
#define H2       1024
#define CODE     256
#define NLAYERS  5
#define NGRAPHS  2048
#define BN_EPS   1e-5f

typedef __attribute__((ext_vector_type(16))) _Float16 v16h;
typedef __attribute__((ext_vector_type(8)))  _Float16 v8h;
typedef __attribute__((ext_vector_type(8)))  float    v8f;

// ------------------------------------------------------------------
// elementwise utilities
// ------------------------------------------------------------------
__global__ void k_zero(float* __restrict__ p, size_t n) {
  size_t i = (size_t)blockIdx.x * blockDim.x + threadIdx.x;
  size_t stride = (size_t)gridDim.x * blockDim.x;
  for (; i < n; i += stride) p[i] = 0.f;
}

__global__ void k_f32_to_f16(const float* __restrict__ in, _Float16* __restrict__ out, size_t n) {
  size_t i = (size_t)blockIdx.x * blockDim.x + threadIdx.x;
  size_t stride = (size_t)gridDim.x * blockDim.x;
  for (; i < n; i += stride) out[i] = (_Float16)in[i];
}

// W [L][R][C] f32  ->  WT [L][C][R] f16   (per-layer transpose + downconvert)
__global__ void k_transpose_f16(const float* __restrict__ W, _Float16* __restrict__ WT,
                                int L, int R, int C) {
  size_t n = (size_t)L * R * C;
  size_t i = (size_t)blockIdx.x * blockDim.x + threadIdx.x;
  size_t stride = (size_t)gridDim.x * blockDim.x;
  for (; i < n; i += stride) {
    size_t l = i / ((size_t)R * C);
    size_t rem = i - l * (size_t)R * C;
    size_t r = rem / C;
    size_t c = rem - r * C;
    WT[l * (size_t)R * C + c * R + r] = (_Float16)W[i];
  }
}

// ------------------------------------------------------------------
// node embedding init: h[n][f] = x_emb1[x[n,0]][f] + x_emb2[x[n,1]][f]
// ------------------------------------------------------------------
__global__ void k_embed(const int* __restrict__ x, const float* __restrict__ e1,
                        const float* __restrict__ e2, float* __restrict__ h, int N) {
  size_t n = (size_t)N * HIDDEN;
  size_t i = (size_t)blockIdx.x * blockDim.x + threadIdx.x;
  size_t stride = (size_t)gridDim.x * blockDim.x;
  for (; i < n; i += stride) {
    size_t node = i >> 9;
    int f = (int)(i & (HIDDEN - 1));
    h[i] = e1[(size_t)x[2 * node] * HIDDEN + f] + e2[(size_t)x[2 * node + 1] * HIDDEN + f];
  }
}

// self-loop init: agg[i] = h[i] + ee1[4][f] + ee2[0][f]
__global__ void k_agg_init(const float* __restrict__ h, const float* __restrict__ ee1,
                           const float* __restrict__ ee2, float* __restrict__ agg, int N) {
  size_t n = (size_t)N * HIDDEN;
  size_t i = (size_t)blockIdx.x * blockDim.x + threadIdx.x;
  size_t stride = (size_t)gridDim.x * blockDim.x;
  for (; i < n; i += stride) {
    int f = (int)(i & (HIDDEN - 1));
    agg[i] = h[i] + ee1[4 * HIDDEN + f] + ee2[0 * HIDDEN + f];
  }
}

// scatter-add: agg[dst] += h[src] + edge_emb1[attr0] + edge_emb2[attr1]
__global__ __launch_bounds__(256) void k_scatter(const float* __restrict__ h,
    const int* __restrict__ ei, const int* __restrict__ ea,
    const float* __restrict__ ee1, const float* __restrict__ ee2,
    float* __restrict__ agg, int E) {
  int e = blockIdx.x;
  if (e >= E) return;
  int src = ei[e];
  int dst = ei[E + e];
  int a0  = ea[2 * e];
  int a1  = ea[2 * e + 1];
  const float* hs = h + (size_t)src * HIDDEN;
  const float* p1 = ee1 + (size_t)a0 * HIDDEN;
  const float* p2 = ee2 + (size_t)a1 * HIDDEN;
  float* ad = agg + (size_t)dst * HIDDEN;
#pragma unroll
  for (int j = 0; j < HIDDEN / 256; ++j) {
    int f = threadIdx.x + j * 256;
    atomicAdd(&ad[f], hs[f] + p1[f] + p2[f]);
  }
}

// ------------------------------------------------------------------
// WMMA GEMM:  C[M,Ncols] = epilogue(A[M,K] @ W[K,Ncols] + bias)
// A f16 row-major, WT f16 [Ncols,K] (W transposed) row-major.
// Each wave computes a 32x32 output tile (2 A frags x 2 B frags -> 4 WMMAs
// per K-step of 32), 4 waves/block covering 32(M) x 128(N) of C.
// ------------------------------------------------------------------
template<bool RELU, bool OUT_HALF>
__global__ __launch_bounds__(128) void k_gemm(const _Float16* __restrict__ A,
    const _Float16* __restrict__ WT, const float* __restrict__ bias,
    float* __restrict__ Cf, _Float16* __restrict__ Ch, int M, int K, int Ncols) {
  const int lane = threadIdx.x & 31;
  const int wave = threadIdx.x >> 5;
  const int hh   = lane >> 4;       // wave half
  const int r    = lane & 15;
  const int n0   = (blockIdx.x * 4 + wave) * 32;
  const int m0   = blockIdx.y * 32;

  int rowA0 = m0 + r;       if (rowA0 > M - 1) rowA0 = M - 1;  // clamp keeps EXEC uniform
  int rowA1 = m0 + 16 + r;  if (rowA1 > M - 1) rowA1 = M - 1;
  const _Float16* pa0 = A  + (size_t)rowA0 * K + 8 * hh;
  const _Float16* pa1 = A  + (size_t)rowA1 * K + 8 * hh;
  const _Float16* pb0 = WT + (size_t)(n0 + r) * K + 8 * hh;
  const _Float16* pb1 = WT + (size_t)(n0 + 16 + r) * K + 8 * hh;

  v8f acc00 = {}, acc01 = {}, acc10 = {}, acc11 = {};
  for (int k = 0; k < K; k += 32) {
    union { v16h v; v8h p[2]; } a0, a1, b0, b1;
    // A 16x32 f16 layout: lane<16 -> K {k..k+7, k+16..k+23}; lane>=16 -> +8
    a0.p[0] = *(const v8h*)(pa0 + k);
    a0.p[1] = *(const v8h*)(pa0 + k + 16);
    a1.p[0] = *(const v8h*)(pa1 + k);
    a1.p[1] = *(const v8h*)(pa1 + k + 16);
    // B 32x16 f16 layout: lane<16 -> K {k..k+7, k+16..k+23}; lane>=16 -> +8
    b0.p[0] = *(const v8h*)(pb0 + k);
    b0.p[1] = *(const v8h*)(pb0 + k + 16);
    b1.p[0] = *(const v8h*)(pb1 + k);
    b1.p[1] = *(const v8h*)(pb1 + k + 16);
    acc00 = __builtin_amdgcn_wmma_f32_16x16x32_f16(false, a0.v, false, b0.v,
                                                   (short)0, acc00, false, false);
    acc01 = __builtin_amdgcn_wmma_f32_16x16x32_f16(false, a0.v, false, b1.v,
                                                   (short)0, acc01, false, false);
    acc10 = __builtin_amdgcn_wmma_f32_16x16x32_f16(false, a1.v, false, b0.v,
                                                   (short)0, acc10, false, false);
    acc11 = __builtin_amdgcn_wmma_f32_16x16x32_f16(false, a1.v, false, b1.v,
                                                   (short)0, acc11, false, false);
  }

  const int col0 = n0 + r;
  const int col1 = n0 + 16 + r;
  const float bv0 = bias[col0];
  const float bv1 = bias[col1];

  auto store_tile = [&](const v8f& acc, int mbase, int colC, float bv) {
#pragma unroll
    for (int v = 0; v < 8; ++v) {
      int rowC = mbase + v + 8 * hh;   // D layout: VGPR v -> M = v (+8 for upper half)
      if (rowC < M) {
        float y = acc[v] + bv;
        if (RELU) y = fmaxf(y, 0.f);
        if (OUT_HALF) Ch[(size_t)rowC * Ncols + colC] = (_Float16)y;
        else          Cf[(size_t)rowC * Ncols + colC] = y;
      }
    }
  };
  store_tile(acc00, m0,      col0, bv0);
  store_tile(acc01, m0,      col1, bv1);
  store_tile(acc10, m0 + 16, col0, bv0);
  store_tile(acc11, m0 + 16, col1, bv1);
}

// ------------------------------------------------------------------
// BatchNorm: per-column stats (biased var) + fused normalize(+ReLU)
// ------------------------------------------------------------------
__global__ __launch_bounds__(256) void k_bn_stats(const float* __restrict__ h,
    float* __restrict__ mean, float* __restrict__ var, int N) {
  __shared__ float ss[256];
  __shared__ float sq[256];
  int c = blockIdx.x;
  float s = 0.f, q = 0.f;
  for (int r = threadIdx.x; r < N; r += 256) {
    float v = h[(size_t)r * HIDDEN + c];
    s += v; q += v * v;
  }
  ss[threadIdx.x] = s; sq[threadIdx.x] = q;
  __syncthreads();
  for (int w = 128; w > 0; w >>= 1) {
    if (threadIdx.x < w) {
      ss[threadIdx.x] += ss[threadIdx.x + w];
      sq[threadIdx.x] += sq[threadIdx.x + w];
    }
    __syncthreads();
  }
  if (threadIdx.x == 0) {
    float m = ss[0] / N;
    mean[c] = m;
    var[c] = sq[0] / N - m * m;
  }
}

__global__ void k_bn_apply(const float* __restrict__ in, const float* __restrict__ mean,
    const float* __restrict__ var, const float* __restrict__ gamma,
    const float* __restrict__ beta, float* __restrict__ out, int relu, size_t n) {
  size_t i = (size_t)blockIdx.x * blockDim.x + threadIdx.x;
  size_t stride = (size_t)gridDim.x * blockDim.x;
  for (; i < n; i += stride) {
    int c = (int)(i & (HIDDEN - 1));
    float y = (in[i] - mean[c]) * rsqrtf(var[c] + BN_EPS) * gamma[c] + beta[c];
    if (relu) y = fmaxf(y, 0.f);
    out[i] = y;
  }
}

// ------------------------------------------------------------------
// global mean pool over graphs
// ------------------------------------------------------------------
__global__ void k_pool_count(const int* __restrict__ batch, float* __restrict__ counts, int N) {
  int n = blockIdx.x * blockDim.x + threadIdx.x;
  if (n < N) atomicAdd(&counts[batch[n]], 1.f);
}

__global__ __launch_bounds__(256) void k_pool_sum(const float* __restrict__ nodes,
    const int* __restrict__ batch, float* __restrict__ out, int N) {
  int n = blockIdx.x;
  if (n >= N) return;
  int g = batch[n];
  int f = threadIdx.x;                       // CODE == 256 == blockDim
  atomicAdd(&out[(size_t)g * CODE + f], nodes[(size_t)n * CODE + f]);
}

__global__ void k_pool_div(float* __restrict__ out, const float* __restrict__ counts, size_t n) {
  size_t i = (size_t)blockIdx.x * blockDim.x + threadIdx.x;
  size_t stride = (size_t)gridDim.x * blockDim.x;
  for (; i < n; i += stride) out[i] = out[i] / fmaxf(counts[i >> 8], 1.f);
}

// ------------------------------------------------------------------
// launcher
// ------------------------------------------------------------------
extern "C" void kernel_launch(void* const* d_in, const int* in_sizes, int n_in,
                              void* d_out, int out_size, void* d_ws, size_t ws_size,
                              hipStream_t stream) {
  const int*   x         = (const int*)d_in[0];
  const int*   edge_idx  = (const int*)d_in[1];
  const int*   edge_attr = (const int*)d_in[2];
  const int*   batch     = (const int*)d_in[3];
  const float* x_emb1    = (const float*)d_in[4];
  const float* x_emb2    = (const float*)d_in[5];
  const float* edge_emb1 = (const float*)d_in[6];   // [5][6][512]
  const float* edge_emb2 = (const float*)d_in[7];   // [5][3][512]
  const float* w1        = (const float*)d_in[8];   // [5][512][1024]
  const float* b1        = (const float*)d_in[9];   // [5][1024]
  const float* w2        = (const float*)d_in[10];  // [5][1024][512]
  const float* b2        = (const float*)d_in[11];  // [5][512]
  const float* gamma     = (const float*)d_in[12];  // [5][512]
  const float* beta      = (const float*)d_in[13];  // [5][512]
  const float* pw1       = (const float*)d_in[14];  // [512][512]
  const float* pb1       = (const float*)d_in[15];
  const float* pw2       = (const float*)d_in[16];  // [512][256]
  const float* pb2       = (const float*)d_in[17];

  const int N = in_sizes[0] / 2;
  const int E = in_sizes[1] / 2;

  // ---- workspace partition ----
  char* ws = (char*)d_ws;
  size_t off = 0;
  auto alloc = [&](size_t bytes) -> char* {
    char* p = ws + off;
    off = (off + bytes + 255) & ~(size_t)255;
    return p;
  };
  float*    hF   = (float*)alloc((size_t)N * HIDDEN * 4);      // node state f32
  float*    tF   = (float*)alloc((size_t)N * HIDDEN * 4);      // agg f32 / pre-BN f32 / out_nodes f32
  _Float16* aH   = (_Float16*)alloc((size_t)N * HIDDEN * 2);   // agg/h f16 (GEMM input)
  char*     big  = alloc((size_t)N * H2 * 2);                  // z f16 [N,1024] / proj hidden f16
  _Float16* zH   = (_Float16*)big;
  _Float16* wt1  = (_Float16*)alloc((size_t)NLAYERS * H2 * HIDDEN * 2);
  _Float16* wt2  = (_Float16*)alloc((size_t)NLAYERS * HIDDEN * H2 * 2);
  _Float16* pwt1 = (_Float16*)alloc((size_t)HIDDEN * HIDDEN * 2);
  _Float16* pwt2 = (_Float16*)alloc((size_t)CODE * HIDDEN * 2);
  float*    mean = (float*)alloc(HIDDEN * 4);
  float*    var  = (float*)alloc(HIDDEN * 4);
  float*    cnts = (float*)alloc(NGRAPHS * 4);

  const size_t NH = (size_t)N * HIDDEN;
  const int EW = 256;
  const int GSB = 16384;   // grid-stride blocks for elementwise

  // ---- weight prep (transpose + f16) ----
  k_transpose_f16<<<4096, EW, 0, stream>>>(w1, wt1, NLAYERS, HIDDEN, H2);
  k_transpose_f16<<<4096, EW, 0, stream>>>(w2, wt2, NLAYERS, H2, HIDDEN);
  k_transpose_f16<<<1024, EW, 0, stream>>>(pw1, pwt1, 1, HIDDEN, HIDDEN);
  k_transpose_f16<<<512,  EW, 0, stream>>>(pw2, pwt2, 1, HIDDEN, CODE);

  // ---- node embedding ----
  k_embed<<<GSB, EW, 0, stream>>>(x, x_emb1, x_emb2, hF, N);

  const int tilesM32 = (N + 31) / 32;

  // ---- GIN layers ----
  for (int l = 0; l < NLAYERS; ++l) {
    const float* ee1 = edge_emb1 + (size_t)l * 6 * HIDDEN;
    const float* ee2 = edge_emb2 + (size_t)l * 3 * HIDDEN;

    k_agg_init<<<GSB, EW, 0, stream>>>(hF, ee1, ee2, tF, N);
    k_scatter<<<E, EW, 0, stream>>>(hF, edge_idx, edge_attr, ee1, ee2, tF, E);
    k_f32_to_f16<<<GSB, EW, 0, stream>>>(tF, aH, NH);

    // z = relu(agg @ w1 + b1)   [N,1024] f16
    k_gemm<true, true><<<dim3(H2 / 128, tilesM32), 128, 0, stream>>>(
        aH, wt1 + (size_t)l * H2 * HIDDEN, b1 + (size_t)l * H2,
        nullptr, zH, N, HIDDEN, H2);
    // pre = z @ w2 + b2         [N,512] f32 (into tF)
    k_gemm<false, false><<<dim3(HIDDEN / 128, tilesM32), 128, 0, stream>>>(
        zH, wt2 + (size_t)l * HIDDEN * H2, b2 + (size_t)l * HIDDEN,
        tF, nullptr, N, H2, HIDDEN);

    k_bn_stats<<<HIDDEN, 256, 0, stream>>>(tF, mean, var, N);
    k_bn_apply<<<GSB, EW, 0, stream>>>(tF, mean, var,
        gamma + (size_t)l * HIDDEN, beta + (size_t)l * HIDDEN,
        hF, (l < NLAYERS - 1) ? 1 : 0, NH);
  }

  // ---- projector ----
  k_f32_to_f16<<<GSB, EW, 0, stream>>>(hF, aH, NH);
  // p = relu(h @ pw1 + pb1)   [N,512] f16  (into zH region, disjoint from aH)
  _Float16* pH = zH;
  k_gemm<true, true><<<dim3(HIDDEN / 128, tilesM32), 128, 0, stream>>>(
      aH, pwt1, pb1, nullptr, pH, N, HIDDEN, HIDDEN);
  // out_nodes = p @ pw2 + pb2  [N,256] f32 (into tF, disjoint from pH)
  float* outNodes = tF;
  k_gemm<false, false><<<dim3(CODE / 128, tilesM32), 128, 0, stream>>>(
      pH, pwt2, pb2, outNodes, nullptr, N, HIDDEN, CODE);

  // ---- mean pool over graphs ----
  k_zero<<<256, EW, 0, stream>>>((float*)d_out, (size_t)NGRAPHS * CODE);
  k_zero<<<8, EW, 0, stream>>>(cnts, NGRAPHS);
  k_pool_count<<<(N + EW - 1) / EW, EW, 0, stream>>>(batch, cnts, N);
  k_pool_sum<<<N, CODE, 0, stream>>>(outNodes, batch, (float*)d_out, N);
  k_pool_div<<<512, EW, 0, stream>>>((float*)d_out, cnts, (size_t)NGRAPHS * CODE);
}
